// VectorQuantizerEMA_52544629899295
// MI455X (gfx1250) — matrix operand order repaired and checked by reading, not compiled, not run on hardware
//
#include <hip/hip_runtime.h>
#include <hip/hip_bf16.h>
#include <math.h>

typedef __attribute__((ext_vector_type(16))) _Float16 v16h;
typedef __attribute__((ext_vector_type(8)))  _Float16 v8h;
typedef __attribute__((ext_vector_type(8)))  float    v8f;

#define KCODES 8192
#define DDIM   256
#define BB     8
#define HW     4096           // 64*64
#define NTOK   32768          // B*H*W
#define DECAY  0.99f
#define EMA_EPS 1e-5f
#define NORM_EPS 1e-12f

// ---- output layout (float element offsets, reference return order) ----
#define O_ZQST 0u                       // (B,D,H,W) 8388608
#define O_IDX  8388608u                 // (B,H,W)   32768  (written as float)
#define O_ZQ   8421376u                 // (B,D,H,W) 8388608
#define O_NEMB 16809984u                // (K,D)     2097152
#define O_NCS  18907136u                // (K,)      8192
#define O_NEA  18915328u                // (K,D)     2097152

// ---- workspace layout (byte offsets) ----
#define W_FLATZ 0ull                    // f32 N*D        = 33554432 B
#define W_ZNORM 33554432ull             // f16 N*D        = 16777216 B
#define W_ENORM 50331648ull             // f16 K*D        =  4194304 B
#define W_IDX   54525952ull             // i32 N          =   131072 B
#define W_CNT   54657024ull             // f32 K          =    32768 B
#define W_ESUM  54689792ull             // f32 K*D        =  8388608 B
#define W_NCS   63078400ull             // f32 K          =    32768 B
#define W_NSUM  63111168ull             // f32 scalar

// =====================================================================
// 1) (B,D,H,W) -> (N,D): write flat_z (f32) and l2-normalized z (f16).
// =====================================================================
__global__ __launch_bounds__(256) void k_flatten_norm(const float* __restrict__ z,
                                                      float* __restrict__ flatz,
                                                      _Float16* __restrict__ znorm) {
    __shared__ float lds[32 * 257];
    __shared__ float part[32][8];
    __shared__ float inv[32];
    const int t = threadIdx.x;
    const int b = blockIdx.x >> 7;            // /128
    const int hwbase = (blockIdx.x & 127) * 32;

    const int dd0 = t >> 5, hh = t & 31;
#pragma unroll 4
    for (int i = 0; i < 32; ++i) {
        int dd = i * 8 + dd0;
        lds[hh * 257 + dd] = z[(size_t)(b * DDIM + dd) * HW + hwbase + hh];
    }
    __syncthreads();

    {
        int r = t >> 3, p = t & 7;
        float ss = 0.f;
        for (int d = p; d < DDIM; d += 8) { float v = lds[r * 257 + d]; ss += v * v; }
        part[r][p] = ss;
    }
    __syncthreads();
    if (t < 32) {
        float ss = 0.f;
#pragma unroll
        for (int p = 0; p < 8; ++p) ss += part[t][p];
        inv[t] = 1.0f / fmaxf(sqrtf(ss), NORM_EPS);
    }
    __syncthreads();

    for (int r = 0; r < 32; ++r) {
        size_t n = (size_t)b * HW + hwbase + r;
        float v = lds[r * 257 + t];
        flatz[n * DDIM + t] = v;
        znorm[n * DDIM + t] = (_Float16)(v * inv[r]);
    }
}

// =====================================================================
// 2) l2-normalize embedding rows -> f16
// =====================================================================
__global__ __launch_bounds__(256) void k_norm_embed(const float* __restrict__ emb,
                                                    _Float16* __restrict__ enorm) {
    __shared__ float red[256];
    const int k = blockIdx.x, t = threadIdx.x;
    float v = emb[(size_t)k * DDIM + t];
    red[t] = v * v;
    __syncthreads();
    for (int s = 128; s > 0; s >>= 1) {
        if (t < s) red[t] += red[t + s];
        __syncthreads();
    }
    float invn = 1.0f / fmaxf(sqrtf(red[0]), NORM_EPS);
    enorm[(size_t)k * DDIM + t] = (_Float16)(v * invn);
}

// =====================================================================
// 3) fused cosine-sim GEMM + argmax.
//    Block = 128 token rows (8 waves x 16-row register A panels).
//    Pipeline stage = 32 codes (two 16-code tiles, 16KB) staged in LDS via
//    global_load_async_to_lds_b128 (ASYNCcnt), double-buffered, shared by
//    all 8 waves. 16 WMMAs per sync region.
// =====================================================================
#define BROW  544                        // padded LDS bytes per B row (32B aligned)
#define NPAIR 256                        // 512 col tiles / 2 per stage

__global__ __launch_bounds__(256) void k_sim_argmax(const _Float16* __restrict__ zn,
                                                    const _Float16* __restrict__ en,
                                                    int* __restrict__ idx_i,
                                                    float* __restrict__ idx_f) {
    __shared__ char btile[2][32 * BROW];   // 2 x 17408 B

    const int t = threadIdx.x;
    const int lane = t & 31;
    const int wave = t >> 5;
    const int lr = lane & 15;
    const int hi = lane >> 4;
    const int rowb = blockIdx.x * 128 + wave * 16;

    // ---- A fragments in registers (contiguous K = e + 16*hi, matching B) ----
    v16h a[8];
    {
        const _Float16* arow = zn + (size_t)(rowb + lr) * DDIM;
#pragma unroll
        for (int kk = 0; kk < 8; ++kk) {
            const _Float16* p = arow + kk * 32 + hi * 16;
            v8h lo = *(const v8h*)p;
            v8h hh = *(const v8h*)(p + 8);
            a[kk] = __builtin_shufflevector(lo, hh, 0, 1, 2, 3, 4, 5, 6, 7,
                                                    8, 9, 10, 11, 12, 13, 14, 15);
        }
    }

    // ---- async B copy: thread copies 32B of tile0 and 32B of tile1 ----
    const int brow = t >> 4;             // 0..15 : code within tile
    const int bchk = t & 15;             // 0..15 : 32B chunk within 512B row
    const unsigned long long gbase =
        (unsigned long long)(uintptr_t)en + (size_t)brow * 512 + (size_t)bchk * 32;
    unsigned int lA[2], lB[2];
    lA[0] = (unsigned int)(uintptr_t)&btile[0][brow * BROW + bchk * 32];
    lB[0] = (unsigned int)(uintptr_t)&btile[0][(brow + 16) * BROW + bchk * 32];
    lA[1] = (unsigned int)(uintptr_t)&btile[1][brow * BROW + bchk * 32];
    lB[1] = (unsigned int)(uintptr_t)&btile[1][(brow + 16) * BROW + bchk * 32];

    // prologue: pair 0 -> buffer 0 (4 async b128 per thread)
    asm volatile("global_load_async_to_lds_b128 %0, %2, off\n\t"
                 "global_load_async_to_lds_b128 %0, %2, off offset:16\n\t"
                 "global_load_async_to_lds_b128 %1, %3, off\n\t"
                 "global_load_async_to_lds_b128 %1, %3, off offset:16"
                 :: "v"(lA[0]), "v"(lB[0]), "v"(gbase), "v"(gbase + 8192ull)
                 : "memory");

    float maxv[8];
    int   maxi[8];
#pragma unroll
    for (int r = 0; r < 8; ++r) { maxv[r] = -3.4e38f; maxi[r] = 0; }

    for (int i = 0; i < NPAIR; ++i) {
        if (i + 1 < NPAIR) {
            unsigned long long g = gbase + (unsigned long long)(i + 1) * 16384ull;
            int nb = (i + 1) & 1;
            asm volatile("global_load_async_to_lds_b128 %0, %2, off\n\t"
                         "global_load_async_to_lds_b128 %0, %2, off offset:16\n\t"
                         "global_load_async_to_lds_b128 %1, %3, off\n\t"
                         "global_load_async_to_lds_b128 %1, %3, off offset:16"
                         :: "v"(lA[nb]), "v"(lB[nb]), "v"(g), "v"(g + 8192ull)
                         : "memory");
            asm volatile("s_wait_asynccnt 0x4" ::: "memory");   // pair i resident
        } else {
            asm volatile("s_wait_asynccnt 0x0" ::: "memory");   // last pair resident
        }
        __syncthreads();

        const _Float16* base = (const _Float16*)&btile[i & 1][0];
#pragma unroll
        for (int s = 0; s < 2; ++s) {
            // B fragment: lane column n = lr, K = e + 16*hi (contiguous v16h)
            const _Float16* bp = base + (s * 16 + lr) * (BROW / 2) + hi * 16;
            v16h bf[8];
#pragma unroll
            for (int kk = 0; kk < 8; ++kk) bf[kk] = *(const v16h*)(bp + kk * 32);
            v8f acc = {};
#pragma unroll
            for (int kk = 0; kk < 8; ++kk) {
                acc = __builtin_amdgcn_wmma_f32_16x16x32_f16(
                    false, a[kk], false, bf[kk], (short)0, acc, false, false);
            }
            // acc[r] = sim[rowb + r + 8*hi][(2i+s)*16 + lr]
            int ci = (i * 2 + s) * 16 + lr;
#pragma unroll
            for (int r = 0; r < 8; ++r) {
                if (acc[r] > maxv[r]) { maxv[r] = acc[r]; maxi[r] = ci; }
            }
        }
        __syncthreads();   // all waves done reading before buffer is overwritten
    }

    // ---- argmax over columns: butterfly across 16 lanes of each half ----
#pragma unroll
    for (int m = 8; m >= 1; m >>= 1) {
#pragma unroll
        for (int r = 0; r < 8; ++r) {
            float ov = __shfl_xor(maxv[r], m, 32);
            int   oi = __shfl_xor(maxi[r], m, 32);
            if (ov > maxv[r] || (ov == maxv[r] && oi < maxi[r])) {
                maxv[r] = ov; maxi[r] = oi;
            }
        }
    }
    if (lr == 0) {
#pragma unroll
        for (int r = 0; r < 8; ++r) {
            int n = rowb + r + 8 * hi;
            idx_i[n] = maxi[r];
            idx_f[n] = (float)maxi[r];
        }
    }
}

// =====================================================================
// 4) z_q gather: out (B,D,H,W); z_q_st == z_q in value.
// =====================================================================
__global__ __launch_bounds__(256) void k_gather_zq(const float* __restrict__ emb,
                                                   const int* __restrict__ idx,
                                                   float* __restrict__ out) {
    const int bd = blockIdx.x >> 4;                       // b*256 + d
    const int hw = ((blockIdx.x & 15) << 8) + threadIdx.x;
    const int b = bd >> 8, d = bd & 255;
    const int n = b * HW + hw;
    int code = idx[n];
    float v = emb[(size_t)code * DDIM + d];
    size_t o = (size_t)bd * HW + hw;
    out[O_ZQST + o] = v;
    out[O_ZQ + o]   = v;
}

// =====================================================================
// 5) zero counts + embed_sum + nsum region of workspace
// =====================================================================
__global__ __launch_bounds__(256) void k_zero(float* __restrict__ p, int n) {
    int stride = gridDim.x * blockDim.x;
    for (int i = blockIdx.x * blockDim.x + threadIdx.x; i < n; i += stride) p[i] = 0.f;
}

// =====================================================================
// 6) scatter-add: counts[idx[n]] += 1 ; embed_sum[idx[n]] += flat_z[n]
// =====================================================================
__global__ __launch_bounds__(256) void k_scatter(const float* __restrict__ flatz,
                                                 const int* __restrict__ idx,
                                                 float* __restrict__ counts,
                                                 float* __restrict__ esum) {
    const int n = blockIdx.x, t = threadIdx.x;
    int c = idx[n];
    atomicAdd(&esum[(size_t)c * DDIM + t], flatz[(size_t)n * DDIM + t]);
    if (t == 0) atomicAdd(&counts[c], 1.0f);
}

// =====================================================================
// 7) new_cluster_size = cs*d + counts*(1-d); global sum n
// =====================================================================
__global__ __launch_bounds__(256) void k_cluster(const float* __restrict__ cs,
                                                 const float* __restrict__ counts,
                                                 float* __restrict__ ncs,
                                                 float* __restrict__ nsum,
                                                 float* __restrict__ out) {
    __shared__ float red[256];
    const int k = blockIdx.x * 256 + threadIdx.x;
    float v = cs[k] * DECAY + counts[k] * (1.0f - DECAY);
    ncs[k] = v;
    out[O_NCS + k] = v;
    red[threadIdx.x] = v;
    __syncthreads();
    for (int s = 128; s > 0; s >>= 1) {
        if (threadIdx.x < s) red[threadIdx.x] += red[threadIdx.x + s];
        __syncthreads();
    }
    if (threadIdx.x == 0) atomicAdd(nsum, red[0]);
}

// =====================================================================
// 8) new_embed_avg + smoothed counts + l2 renorm -> new_embedding
// =====================================================================
__global__ __launch_bounds__(256) void k_finalize(const float* __restrict__ ea,
                                                  const float* __restrict__ esum,
                                                  const float* __restrict__ ncs,
                                                  const float* __restrict__ nsum,
                                                  float* __restrict__ out) {
    __shared__ float red[256];
    const int k = blockIdx.x, t = threadIdx.x;
    float nall = nsum[0];
    float csk = (ncs[k] + EMA_EPS) / (nall + (float)KCODES * EMA_EPS) * nall;
    float av = ea[(size_t)k * DDIM + t] * DECAY + esum[(size_t)k * DDIM + t] * (1.0f - DECAY);
    out[O_NEA + (size_t)k * DDIM + t] = av;
    float v = av / csk;
    red[t] = v * v;
    __syncthreads();
    for (int s = 128; s > 0; s >>= 1) {
        if (t < s) red[t] += red[t + s];
        __syncthreads();
    }
    float invn = 1.0f / fmaxf(sqrtf(red[0]), NORM_EPS);
    out[O_NEMB + (size_t)k * DDIM + t] = v * invn;
}

// =====================================================================
extern "C" void kernel_launch(void* const* d_in, const int* in_sizes, int n_in,
                              void* d_out, int out_size, void* d_ws, size_t ws_size,
                              hipStream_t stream) {
    const float* z   = (const float*)d_in[0];   // (B,D,H,W)
    const float* emb = (const float*)d_in[1];   // (K,D)
    const float* cs  = (const float*)d_in[2];   // (K,)
    const float* ea  = (const float*)d_in[3];   // (K,D)
    float* out = (float*)d_out;

    char* ws = (char*)d_ws;
    float*    flatz  = (float*)(ws + W_FLATZ);
    _Float16* znorm  = (_Float16*)(ws + W_ZNORM);
    _Float16* enorm  = (_Float16*)(ws + W_ENORM);
    int*      idx_i  = (int*)(ws + W_IDX);
    float*    counts = (float*)(ws + W_CNT);
    float*    esum   = (float*)(ws + W_ESUM);
    float*    ncs    = (float*)(ws + W_NCS);
    float*    nsum   = (float*)(ws + W_NSUM);

    // 1) flatten + normalize z
    k_flatten_norm<<<BB * (HW / 32), 256, 0, stream>>>(z, flatz, znorm);
    // 2) normalize embedding
    k_norm_embed<<<KCODES, 256, 0, stream>>>(emb, enorm);
    // 5) zero counts/esum/nsum (contiguous region)
    {
        int nz = (int)((W_NSUM + 16 - W_CNT) / 4);
        k_zero<<<2048, 256, 0, stream>>>(counts, nz);
    }
    // 3) fused WMMA GEMM + argmax : N/128 = 256 blocks
    k_sim_argmax<<<NTOK / 128, 256, 0, stream>>>(znorm, enorm, idx_i, out + O_IDX);
    // 4) gather z_q / z_q_st
    k_gather_zq<<<BB * DDIM * (HW / 256), 256, 0, stream>>>(emb, idx_i, out);
    // 6) scatter-add EMA statistics
    k_scatter<<<NTOK, 256, 0, stream>>>(flatz, idx_i, counts, esum);
    // 7) new cluster size + total n
    k_cluster<<<KCODES / 256, 256, 0, stream>>>(cs, counts, ncs, nsum, out);
    // 8) new embed_avg + renormalized embedding
    k_finalize<<<KCODES, 256, 0, stream>>>(ea, esum, ncs, nsum, out);
}
